// Memory_2362232013133
// MI455X (gfx1250) — matrix operand order repaired and checked by reading, not compiled, not run on hardware
//
#include <hip/hip_runtime.h>
#include <hip/hip_bf16.h>

// CDNA5 / gfx1250, wave32. bf16 WMMA (V_WMMA_F32_16X16X32_BF16), f32 accum.
// Data movement showcases both gfx1250 DMA paths:
//   - summary GEMM: global_load_async_to_lds_b128 + s_wait_asynccnt (ASYNCcnt)
//   - qk/qv GEMM:   TENSOR_LOAD_TO_LDS (TDM descriptor, TENSORcnt) w/ fallback
// Dead code in reference: sa_wq/sa_wk/sa_wv branch never reaches the output.

typedef __attribute__((ext_vector_type(16))) __bf16 v16bf;
typedef __attribute__((ext_vector_type(8)))  float  v8f;
typedef __attribute__((ext_vector_type(4)))  unsigned int u32x4;
typedef __attribute__((ext_vector_type(8)))  int i32x8;
typedef __attribute__((ext_vector_type(4)))  int i32x4;

#define WMMA_BF16(A, B, C) \
  __builtin_amdgcn_wmma_f32_16x16x32_bf16(false, (A), false, (B), (short)0, (C), false, false)

__device__ __forceinline__ void cvt4(v16bf& v, int base, float4 f) {
  v[base + 0] = (__bf16)f.x;
  v[base + 1] = (__bf16)f.y;
  v[base + 2] = (__bf16)f.z;
  v[base + 3] = (__bf16)f.w;
}

// 16 consecutive fp32 (16B aligned) -> bf16 B-fragment lane slice.
__device__ __forceinline__ v16bf load_b16(const float* p) {
  const float4* p4 = reinterpret_cast<const float4*>(p);
  v16bf v;
  cvt4(v, 0, p4[0]); cvt4(v, 4, p4[1]); cvt4(v, 8, p4[2]); cvt4(v, 12, p4[3]);
  return v;
}

// A-matrix 16x32 bf16 lane slice: elems 0..15 = K[+0..+15]
// relative to (k0 + kg*16) (contiguous k order, matches B fragments).
__device__ __forceinline__ v16bf load_a16(const float* row_k0_kg8) {
  const float4* p4 = reinterpret_cast<const float4*>(row_k0_kg8);
  v16bf v;
  cvt4(v, 0, p4[0]); cvt4(v, 4, p4[1]);
  cvt4(v, 8, p4[2]); cvt4(v, 12, p4[3]);
  return v;
}

// --- gfx1250 async copy: 16B global -> LDS, tracked by ASYNCcnt ------------
__device__ __forceinline__ void async_copy16(const float* g, unsigned lds_off) {
  asm volatile("global_load_async_to_lds_b128 %0, %1, off"
               :: "v"(lds_off), "v"((unsigned long long)(uintptr_t)g)
               : "memory");
}
__device__ __forceinline__ void wait_async0() {
  asm volatile("s_wait_asynccnt 0x0" ::: "memory");
}

// LDS B-strip layout: 64 rows (output cols) x 32 k, fp32, row stride 36 floats
// (144B: 16B-aligned for b128, conflict-free 16-lane bank pattern).
#define BSTRIDE 36

// --- gfx1250 TDM: one descriptor stages the whole 64x32 fp32 B tile --------
#if __has_builtin(__builtin_amdgcn_tensor_load_to_lds)
#define HAVE_TDM 1
// D# per CDNA5 ISA: 2D tile, data_size=4B,
// tile 32(elems) x 64(rows), tensor_dim0_stride=K, LDS padding 4 DWORDs after
// every 32 DWORDs -> 144B row stride (matches BSTRIDE).
__device__ __forceinline__ void tdm_stage_b(const float* gbase, unsigned lds_addr,
                                            int K) {
  const unsigned long long ga = (unsigned long long)(uintptr_t)gbase;
  u32x4 g0;
  g0[0] = 1u;                                                  // count=1 (valid)
  g0[1] = lds_addr;                                            // lds_addr[31:0]
  g0[2] = (unsigned)ga;                                        // global_addr lo
  g0[3] = ((unsigned)(ga >> 32) & 0x01FFFFFFu) | 0x80000000u;  // addr hi | type=2
  const unsigned td0 = (unsigned)K;
  i32x8 g1;
  g1[0] = (int)((2u << 16)      // data_size = 4B
              | (1u << 20)      // pad_enable
              | (4u << 22)      // pad_interval: every 32 DWORDs
              | (3u << 25));    // pad_amount: 4 DWORDs
  g1[1] = (int)((td0 & 0xFFFFu) << 16);              // tensor_dim0[15:0]
  g1[2] = (int)((td0 >> 16) | (64u << 16));          // dim0 hi | tensor_dim1=64
  g1[3] = (int)(32u << 16);                          // tile_dim0 = 32 elems
  g1[4] = 64;                                        // tile_dim1 = 64
  g1[5] = (int)td0;                                  // dim0_stride lo = K
  g1[6] = 0;                                         // stride hi / dim1_stride
  g1[7] = 0;
  i32x4 z4 = {};
#if __clang_major__ >= 23
  i32x8 z8 = {};
  __builtin_amdgcn_tensor_load_to_lds(g0, g1, z4, z4, z8, 0);
#else
  __builtin_amdgcn_tensor_load_to_lds(g0, g1, z4, z4, 0);
#endif
}
#endif

// ---------------------------------------------------------------------------
// Kernel 1: summary[m,n] = sum_k windows[m,k] * ws_w[n,k]
//   M=2048 (b*512+win), N=1024, K=16384; windows[m, t*1024+d] = x[b, win*8+t, d]
//   (zero if win*8+t >= 4096).
// Block: 256 thr = 8 waves, tile 128(M) x 64(N); B double-buffered in LDS via
// ASYNCcnt loads. Grid (16, 16).
// ---------------------------------------------------------------------------
__global__ void summary_gemm_staged(const float* __restrict__ x,
                                    const float* __restrict__ wsw,
                                    float* __restrict__ out) {
  __shared__ __align__(16) float bbuf[2][64 * BSTRIDE];
  const int tid  = threadIdx.x;
  const int lane = tid & 31;
  const int lr   = lane & 15;
  const int kg   = lane >> 4;
  const int wid  = tid >> 5;
  const int m0 = blockIdx.x * 128 + wid * 16;
  const int n0 = blockIdx.y * 64;

  const int m   = m0 + lr;           // A row for this lane
  const int b   = m >> 9;
  const int win = m & 511;

  // staging assignment: thread -> (row i of 64, 16B chunk c of 4)
  const int srow = tid >> 2, schk = tid & 3;
  const float* sg = wsw + (size_t)(n0 + srow) * 16384 + schk * 4;
  const unsigned soff = (unsigned)(srow * (BSTRIDE * 4) + schk * 16);
  const unsigned lds0 = (unsigned)(uintptr_t)&bbuf[0][0];
  const unsigned lds1 = (unsigned)(uintptr_t)&bbuf[1][0];

  v8f acc0 = {}, acc1 = {}, acc2 = {}, acc3 = {};

  async_copy16(sg, lds0 + soff);  // prologue: stage k0 = 0

#pragma unroll 1
  for (int k0 = 0; k0 < 16384; k0 += 32) {
    const int cur = (k0 >> 5) & 1;
    wait_async0();
    __syncthreads();  // staged B visible to all waves; prev buffer free
    if (k0 + 32 < 16384)
      async_copy16(sg + k0 + 32, (cur ? lds0 : lds1) + soff);

    // A fragment: strided window gather (zero-pad past row 4095)
    const int t      = k0 >> 10;
    const int d0     = k0 & 1023;
    const int seqrow = win * 8 + t;
    v16bf a;
    if (seqrow < 4096) {
      a = load_a16(x + ((size_t)b * 4096 + seqrow) * 1024 + d0 + kg * 16);
    } else {
#pragma unroll
      for (int j = 0; j < 16; ++j) a[j] = (__bf16)0.f;
    }

    const float* bb = &bbuf[cur][kg * 16];
    v16bf f0 = load_b16(bb + (0 * 16 + lr) * BSTRIDE);
    v16bf f1 = load_b16(bb + (1 * 16 + lr) * BSTRIDE);
    v16bf f2 = load_b16(bb + (2 * 16 + lr) * BSTRIDE);
    v16bf f3 = load_b16(bb + (3 * 16 + lr) * BSTRIDE);

    acc0 = WMMA_BF16(a, f0, acc0);
    acc1 = WMMA_BF16(a, f1, acc1);
    acc2 = WMMA_BF16(a, f2, acc2);
    acc3 = WMMA_BF16(a, f3, acc3);
  }

#pragma unroll
  for (int r = 0; r < 8; ++r) {
    const size_t o = (size_t)(m0 + kg * 8 + r) * 1024;
    out[o + n0 + lr]      = acc0[r];
    out[o + n0 + 16 + lr] = acc1[r];
    out[o + n0 + 32 + lr] = acc2[r];
    out[o + n0 + 48 + lr] = acc3[r];
  }
}

// ---------------------------------------------------------------------------
// Kernel 2: staged C = A(MxK) @ B(NxK)^T, fp32 in/out, bf16 compute.
// Block tile 128(M) x 64(N), double-buffered B staged by the TDM (one
// descriptor per k-step, issued by wave 0, TENSORcnt) or by ASYNCcnt loads.
// Grid (M/128, N/64). Requires M%128==0, N%64==0, K%32==0.
// ---------------------------------------------------------------------------
__global__ void gemm_nt_staged(const float* __restrict__ A, const float* __restrict__ B,
                               float* __restrict__ C, int M, int N, int K) {
  __shared__ __align__(16) float bbuf[2][64 * BSTRIDE];
  const int tid  = threadIdx.x;
  const int lane = tid & 31;
  const int lr   = lane & 15;
  const int kg   = lane >> 4;
  const int wid  = tid >> 5;
  const int m0 = blockIdx.x * 128 + wid * 16;
  const int n0 = blockIdx.y * 64;

  const unsigned lds0 = (unsigned)(uintptr_t)&bbuf[0][0];
  const unsigned lds1 = (unsigned)(uintptr_t)&bbuf[1][0];

#ifdef HAVE_TDM
  const float* bstrip = B + (size_t)n0 * K;    // tile start; dim0 stride = K
#else
  const int srow = tid >> 2, schk = tid & 3;
  const float* sg = B + (size_t)(n0 + srow) * K + schk * 4;
  const unsigned soff = (unsigned)(srow * (BSTRIDE * 4) + schk * 16);
#endif

  const float* arow = A + (size_t)(m0 + lr) * K;

  v8f acc0 = {}, acc1 = {}, acc2 = {}, acc3 = {};

#ifdef HAVE_TDM
  if (wid == 0) tdm_stage_b(bstrip, lds0, K);  // prologue: stage k0 = 0
#else
  async_copy16(sg, lds0 + soff);
#endif

#pragma unroll 1
  for (int k0 = 0; k0 < K; k0 += 32) {
    const int cur = (k0 >> 5) & 1;
#ifdef HAVE_TDM
    if (wid == 0) __builtin_amdgcn_s_wait_tensorcnt((unsigned short)0);
    __syncthreads();  // staged B visible to all waves; prev buffer free
    if (k0 + 32 < K && wid == 0)
      tdm_stage_b(bstrip + k0 + 32, cur ? lds0 : lds1, K);
#else
    wait_async0();
    __syncthreads();
    if (k0 + 32 < K)
      async_copy16(sg + k0 + 32, (cur ? lds0 : lds1) + soff);
#endif

    v16bf a = load_a16(arow + k0 + kg * 16);
    const float* bb = &bbuf[cur][kg * 16];
    v16bf f0 = load_b16(bb + (0 * 16 + lr) * BSTRIDE);
    v16bf f1 = load_b16(bb + (1 * 16 + lr) * BSTRIDE);
    v16bf f2 = load_b16(bb + (2 * 16 + lr) * BSTRIDE);
    v16bf f3 = load_b16(bb + (3 * 16 + lr) * BSTRIDE);

    acc0 = WMMA_BF16(a, f0, acc0);
    acc1 = WMMA_BF16(a, f1, acc1);
    acc2 = WMMA_BF16(a, f2, acc2);
    acc3 = WMMA_BF16(a, f3, acc3);
  }

#pragma unroll
  for (int r = 0; r < 8; ++r) {
    const size_t o = (size_t)(m0 + kg * 8 + r) * N;
    C[o + n0 + lr]      = acc0[r];
    C[o + n0 + 16 + lr] = acc1[r];
    C[o + n0 + 32 + lr] = acc2[r];
    C[o + n0 + 48 + lr] = acc3[r];
  }
}

// ---------------------------------------------------------------------------
// Kernel 3: small per-wave C = alpha * A(MxK) @ B(NxK)^T, batched (blockIdx.y).
// Used only for the tiny logits GEMM (M=64, N=512, K=1024 per batch).
// ---------------------------------------------------------------------------
__global__ void gemm_bf16_nt(const float* __restrict__ A, const float* __restrict__ B,
                             float* __restrict__ C, int M, int N, int K,
                             long long sAb, long long sBb, long long sCb, float alpha) {
  const int lane = threadIdx.x & 31;
  const int lr   = lane & 15;
  const int kg   = lane >> 4;
  const int tile   = blockIdx.x * (blockDim.x >> 5) + (threadIdx.x >> 5);
  const int tilesN = N >> 6;
  const int tile_m = tile / tilesN;
  const int tile_n = tile - tile_m * tilesN;
  const int m0 = tile_m << 4;
  const int n0 = tile_n << 6;
  if (m0 >= M) return;  // wave-uniform guard

  const float* Ab = A + (size_t)blockIdx.y * sAb;
  const float* Bb = B + (size_t)blockIdx.y * sBb;
  float*       Cb = C + (size_t)blockIdx.y * sCb;

  const float* arow = Ab + (size_t)(m0 + lr) * K;
  const float* br0  = Bb + (size_t)(n0 + lr) * K;
  const float* br1  = br0 + (size_t)16 * K;
  const float* br2  = br0 + (size_t)32 * K;
  const float* br3  = br0 + (size_t)48 * K;

  v8f acc0 = {}, acc1 = {}, acc2 = {}, acc3 = {};

#pragma unroll 1
  for (int k0 = 0; k0 < K; k0 += 32) {
    v16bf a = load_a16(arow + k0 + kg * 16);
    const int kb = k0 + kg * 16;
    v16bf f0 = load_b16(br0 + kb);
    v16bf f1 = load_b16(br1 + kb);
    v16bf f2 = load_b16(br2 + kb);
    v16bf f3 = load_b16(br3 + kb);
    acc0 = WMMA_BF16(a, f0, acc0);
    acc1 = WMMA_BF16(a, f1, acc1);
    acc2 = WMMA_BF16(a, f2, acc2);
    acc3 = WMMA_BF16(a, f3, acc3);
  }

#pragma unroll
  for (int r = 0; r < 8; ++r) {
    const size_t o = (size_t)(m0 + kg * 8 + r) * N;
    Cb[o + n0 + lr]      = alpha * acc0[r];
    Cb[o + n0 + 16 + lr] = alpha * acc1[r];
    Cb[o + n0 + 32 + lr] = alpha * acc2[r];
    Cb[o + n0 + 48 + lr] = alpha * acc3[r];
  }
}

// ---------------------------------------------------------------------------
// Kernel 4: C = A(MxK) @ B(KxN), B row-major (ldb), batched.
// Used for memory = p @ qv (M=64, N=1024, K=512, ldb=1024).
// ---------------------------------------------------------------------------
__global__ void gemm_bf16_nn(const float* __restrict__ A, const float* __restrict__ B,
                             float* __restrict__ C, int M, int N, int K, int ldb,
                             long long sAb, long long sBb, long long sCb) {
  const int lane = threadIdx.x & 31;
  const int lr   = lane & 15;
  const int kg   = lane >> 4;
  const int tile   = blockIdx.x * (blockDim.x >> 5) + (threadIdx.x >> 5);
  const int tilesN = N >> 6;
  const int tile_m = tile / tilesN;
  const int tile_n = tile - tile_m * tilesN;
  const int m0 = tile_m << 4;
  const int n0 = tile_n << 6;
  if (m0 >= M) return;

  const float* Ab = A + (size_t)blockIdx.y * sAb;
  const float* Bb = B + (size_t)blockIdx.y * sBb;
  float*       Cb = C + (size_t)blockIdx.y * sCb;

  const float* arow = Ab + (size_t)(m0 + lr) * K;
  const int col0 = n0 + lr;

  v8f acc0 = {}, acc1 = {}, acc2 = {}, acc3 = {};

#pragma unroll 1
  for (int k0 = 0; k0 < K; k0 += 32) {
    v16bf a = load_a16(arow + k0 + kg * 16);
    const int kb = k0 + kg * 16;
    v16bf f0, f1, f2, f3;
#pragma unroll
    for (int e = 0; e < 16; ++e) {
      const float* brow = Bb + (size_t)(kb + e) * ldb;
      f0[e] = (__bf16)brow[col0];
      f1[e] = (__bf16)brow[col0 + 16];
      f2[e] = (__bf16)brow[col0 + 32];
      f3[e] = (__bf16)brow[col0 + 48];
    }
    acc0 = WMMA_BF16(a, f0, acc0);
    acc1 = WMMA_BF16(a, f1, acc1);
    acc2 = WMMA_BF16(a, f2, acc2);
    acc3 = WMMA_BF16(a, f3, acc3);
  }

#pragma unroll
  for (int r = 0; r < 8; ++r) {
    const size_t o = (size_t)(m0 + kg * 8 + r) * N;
    Cb[o + n0 + lr]      = acc0[r];
    Cb[o + n0 + 16 + lr] = acc1[r];
    Cb[o + n0 + 32 + lr] = acc2[r];
    Cb[o + n0 + 48 + lr] = acc3[r];
  }
}

// ---------------------------------------------------------------------------
// Kernel 5: in-place row softmax, one wave32 per row.
// ---------------------------------------------------------------------------
__global__ void softmax_rows(float* __restrict__ p, int cols) {
  const int lane = threadIdx.x & 31;
  const int row  = blockIdx.x * (blockDim.x >> 5) + (threadIdx.x >> 5);
  float* r = p + (size_t)row * cols;

  float mx = -3.4e38f;
  for (int c = lane; c < cols; c += 32) mx = fmaxf(mx, r[c]);
#pragma unroll
  for (int off = 16; off; off >>= 1) mx = fmaxf(mx, __shfl_xor(mx, off, 32));

  float s = 0.f;
  for (int c = lane; c < cols; c += 32) {
    float e = __expf(r[c] - mx);
    r[c] = e;
    s += e;
  }
#pragma unroll
  for (int off = 16; off; off >>= 1) s += __shfl_xor(s, off, 32);

  const float inv = 1.f / s;
  for (int c = lane; c < cols; c += 32) r[c] *= inv;
}

// ---------------------------------------------------------------------------
// Kernel 6: per-batch double RMS-norm (means over the whole 64x1024 slab).
// ---------------------------------------------------------------------------
__device__ __forceinline__ float block_reduce(float v, float* sh) {
#pragma unroll
  for (int off = 16; off; off >>= 1) v += __shfl_xor(v, off, 32);
  const int wid = threadIdx.x >> 5, lane = threadIdx.x & 31;
  if (lane == 0) sh[wid] = v;
  __syncthreads();
  if (wid == 0) {
    float t = sh[lane];  // blockDim == 1024 -> exactly 32 partials
#pragma unroll
    for (int off = 16; off; off >>= 1) t += __shfl_xor(t, off, 32);
    if (lane == 0) sh[0] = t;
  }
  __syncthreads();
  const float r = sh[0];
  __syncthreads();
  return r;
}

__global__ void fuse_norm(const float* __restrict__ mem, const float* __restrict__ h,
                          const float* __restrict__ mn_w, const float* __restrict__ hn_w,
                          float* __restrict__ out) {
  const int tid = threadIdx.x;
  const size_t base = (size_t)blockIdx.x * 65536;
  __shared__ float sh[32];
  const float eps = 1.1920929e-07f;
  const float inv_n = 1.f / 65536.f;

  float s = 0.f;
  for (int i = tid; i < 65536; i += 1024) { float v = mem[base + i]; s += v * v; }
  const float r1 = rsqrtf(block_reduce(s, sh) * inv_n + eps);

  float s2 = 0.f;
  for (int i = tid; i < 65536; i += 1024) {
    float hv = h[base + i] + mem[base + i] * r1 * mn_w[i];
    s2 += hv * hv;
  }
  const float r2 = rsqrtf(block_reduce(s2, sh) * inv_n + eps);

  for (int i = tid; i < 65536; i += 1024) {
    float hv = h[base + i] + mem[base + i] * r1 * mn_w[i];
    out[base + i] = hv * r2 * hn_w[i];
  }
}

// ---------------------------------------------------------------------------
extern "C" void kernel_launch(void* const* d_in, const int* in_sizes, int n_in,
                              void* d_out, int out_size, void* d_ws, size_t ws_size,
                              hipStream_t stream) {
  (void)in_sizes; (void)n_in; (void)out_size; (void)ws_size;
  const float* x     = (const float*)d_in[0];   // (4,4096,1024)
  const float* h     = (const float*)d_in[1];   // (4,64,1024)
  const float* ws_w  = (const float*)d_in[2];   // (1024,16384)
  // d_in[3..5] = sa_wq / sa_wk / sa_wv : dead code in reference, unused.
  const float* qa_q  = (const float*)d_in[6];   // (64,1024)
  const float* qa_wk = (const float*)d_in[7];   // (1024,1024)
  const float* qa_wv = (const float*)d_in[8];   // (1024,1024)
  const float* mn_w  = (const float*)d_in[9];   // (64,1024)
  const float* hn_w  = (const float*)d_in[10];  // (64,1024)
  float* out = (float*)d_out;                   // (4,64,1024)

  float* ws      = (float*)d_ws;
  float* summary = ws;                          // 2048*1024
  float* qk      = summary + 2048 * 1024;       // 2048*1024
  float* qv      = qk + 2048 * 1024;            // 2048*1024
  float* probs   = qv + 2048 * 1024;            // 4*64*512
  float* memv    = probs + 4 * 64 * 512;        // 4*64*1024  (~26 MB total)

  // 1) summary = windows(x) @ ws_w^T : block tile 128x64, ASYNC-staged B
  summary_gemm_staged<<<dim3(16, 16), 256, 0, stream>>>(x, ws_w, summary);

  // 2) qk / qv = summary @ W^T : M=2048,N=1024,K=1024, TDM-staged B
  gemm_nt_staged<<<dim3(16, 16), 256, 0, stream>>>(summary, qa_wk, qk, 2048, 1024, 1024);
  gemm_nt_staged<<<dim3(16, 16), 256, 0, stream>>>(summary, qa_wv, qv, 2048, 1024, 1024);

  // 3) logits = (qa_q @ qk^T) / sqrt(1024) : per batch M=64,N=512,K=1024
  gemm_bf16_nt<<<dim3(4, 4), 256, 0, stream>>>(qa_q, qk, probs,
                                               64, 512, 1024,
                                               0, (long long)512 * 1024,
                                               (long long)64 * 512, 0.03125f);

  // 4) softmax over n (512) : 256 rows, 8 waves/block
  softmax_rows<<<dim3(32), 256, 0, stream>>>(probs, 512);

  // 5) memory = p @ qv : per batch M=64,N=1024,K=512
  gemm_bf16_nn<<<dim3(8, 4), 256, 0, stream>>>(probs, qv, memv,
                                               64, 1024, 512, 1024,
                                               (long long)64 * 512,
                                               (long long)512 * 1024,
                                               (long long)64 * 1024);

  // 6) h' = h + rmsnorm2d(memory)*mn_w ; out = rmsnorm2d(h')*hn_w
  fuse_norm<<<dim3(4), 1024, 0, stream>>>(memv, h, mn_w, hn_w, out);
}
